// STGAT_50680614093552
// MI455X (gfx1250) — matrix operand order repaired and checked by reading, not compiled, not run on hardware
//
#include <hip/hip_runtime.h>
#include <hip/hip_bf16.h>
#include <math.h>

// ---------------- problem constants ----------------
#define NN    20000
#define EE    400000
#define TT    8
#define CIN   8
#define HID   16
#define NHEAD 4
#define FF    128      // HID*T
#define OD    512      // HEADS*HID*T
#define EPSV  1e-5f

typedef __attribute__((ext_vector_type(16))) _Float16 v16h;
typedef __attribute__((ext_vector_type(8)))  float    v8f;
typedef __attribute__((ext_vector_type(4)))  unsigned u32x4;

// 16-half WMMA fragment viewed as two 16-byte quads (VGPRs 0-3 / 4-7)
union Frag16 { v16h v; u32x4 q[2]; };

// ---------------- small helpers ----------------
__device__ __forceinline__ float sigmoidf_(float x) { return 1.0f / (1.0f + expf(-x)); }
__device__ __forceinline__ float leaky02(float x)   { return x > 0.0f ? x : 0.2f * x; }

// monotonic ordered-float key for unsigned atomicMax
__device__ __forceinline__ unsigned fkey(float f) {
    unsigned u = __float_as_uint(f);
    return (u & 0x80000000u) ? ~u : (u | 0x80000000u);
}
__device__ __forceinline__ float funkey(unsigned u) {
    return (u & 0x80000000u) ? __uint_as_float(u & 0x7fffffffu) : __uint_as_float(~u);
}

// ---------------- generic zero ----------------
__global__ void k_zero(float* p, int n) {
    int i = blockIdx.x * blockDim.x + threadIdx.x;
    if (i < n) p[i] = 0.0f;
}

// ---------------- kernel 1: pointwise-in + residual + gated_conv1 + bn3 stats ----
// one block (128 threads) per node; tid = c*8 + t
__global__ __launch_bounds__(128) void k_node1(
    const float* __restrict__ X,
    const float* __restrict__ w_in, const float* __restrict__ b_in,
    const float* __restrict__ res_w, const float* __restrict__ res_b,
    const float* __restrict__ w1, const float* __restrict__ b1,
    const float* __restrict__ w2, const float* __restrict__ b2,
    const float* __restrict__ w3, const float* __restrict__ b3,
    float* __restrict__ resid, float* __restrict__ h1, float* __restrict__ st1)
{
    __shared__ float Xs[CIN * TT];   // 64
    __shared__ float xs[HID * TT];   // 128
    __shared__ float red[HID * TT];  // 128
    const int n   = blockIdx.x;
    const int tid = threadIdx.x;
    const int c   = tid >> 3;
    const int t   = tid & 7;

    if (tid < CIN * TT) Xs[tid] = X[(size_t)n * (CIN * TT) + tid];
    __syncthreads();

    // x = w_in @ X  (16x8 per node)
    float xv = b_in[c];
    #pragma unroll
    for (int ci = 0; ci < CIN; ++ci) xv += w_in[c * CIN + ci] * Xs[ci * TT + t];
    xs[tid] = xv;
    __syncthreads();

    // residual = res_w @ x
    float rv = res_b[c];
    #pragma unroll
    for (int j = 0; j < HID; ++j) rv += res_w[c * HID + j] * xs[j * TT + t];
    resid[(size_t)n * FF + tid] = rv;

    // gated conv: p * sigmoid(q) + r, relu
    float p = b1[c], q = b2[c], r = b3[c];
    #pragma unroll
    for (int j = 0; j < HID; ++j) {
        const float x0 = (t > 0) ? xs[j * TT + t - 1] : 0.0f;
        const float x1 = xs[j * TT + t];
        const float x2 = (t < 7) ? xs[j * TT + t + 1] : 0.0f;
        const int   wb = (c * HID + j) * 3;
        p += w1[wb] * x0 + w1[wb + 1] * x1 + w1[wb + 2] * x2;
        q += w2[wb] * x0 + w2[wb + 1] * x1 + w2[wb + 2] * x2;
        r += w3[wb] * x0 + w3[wb + 1] * x1 + w3[wb + 2] * x2;
    }
    float h = fmaxf(p * sigmoidf_(q) + r, 0.0f);
    h1[(size_t)n * FF + tid] = h;

    // bn3 stats: reduce over t in LDS, one atomic pair per channel per block
    red[tid] = h;
    __syncthreads();
    if (t == 0) {
        float s = 0.0f, sq = 0.0f;
        #pragma unroll
        for (int tt2 = 0; tt2 < TT; ++tt2) { float v = red[c * TT + tt2]; s += v; sq += v * v; }
        atomicAdd(&st1[c], s);
        atomicAdd(&st1[HID + c], sq);
    }
}

// ---------------- bn3 apply -> f16 A matrix ----------------
__global__ void k_bn3a(const float* __restrict__ h1, const float* __restrict__ st1,
                       _Float16* __restrict__ A16)
{
    int i = blockIdx.x * blockDim.x + threadIdx.x;
    if (i >= NN * FF) return;
    int c = (i >> 3) & (HID - 1);
    const float cnt = (float)(NN * TT);
    float m  = st1[c] / cnt;
    float vr = st1[HID + c] / cnt - m * m;
    A16[i] = (_Float16)((h1[i] - m) * rsqrtf(vr + EPSV));
}

// ---------------- gat_w -> f16, transposed to (OD x FF) for contiguous B frags --
__global__ void k_wcvtT(const float* __restrict__ W, _Float16* __restrict__ WT)
{
    int i = blockIdx.x * blockDim.x + threadIdx.x;  // i indexes W row-major (k*OD + n)
    if (i >= FF * OD) return;
    int k = i >> 9;          // 0..127
    int n = i & (OD - 1);    // 0..511
    WT[(size_t)n * FF + k] = (_Float16)W[i];
}

// ---------------- WMMA GEMM: lin(N x 512) = A16(N x 128) @ W^T(512 x 128) -------
// block = 256 thr (8 waves); block handles 16 rows x 512 cols; wave -> 64 cols
__global__ __launch_bounds__(256) void k_gemm(
    const _Float16* __restrict__ A, const _Float16* __restrict__ BT,
    float* __restrict__ C)
{
    __shared__ __align__(16) _Float16 Ah[16 * FF]; // 4 KB
    const int row0 = blockIdx.x * 16;

    // coalesced A tile load: 16*128 halves = 4 KB = 256 x b128
    {
        const u32x4* Ag = reinterpret_cast<const u32x4*>(A + (size_t)row0 * FF);
        u32x4* As = reinterpret_cast<u32x4*>(Ah);
        As[threadIdx.x] = Ag[threadIdx.x];
    }
    // prefetch next block's A tile (gfx1250 global_prefetch_b8)
    if (blockIdx.x + 1 < gridDim.x)
        __builtin_prefetch(A + (size_t)(row0 + 16) * FF + threadIdx.x * 16, 0, 1);
    __syncthreads();

    const int lane = threadIdx.x & 31;
    const int wave = threadIdx.x >> 5;
    const int m    = lane & 15;
    const int half = lane >> 4;
    const int n    = lane & 15;

    // A fragments (ISA 16-bit A 16x32 layout):
    //   VGPR 0-3: K = kk*32 + half*8 .. +7   (16 contiguous bytes)
    //   VGPR 4-7: K = kk*32 + half*8 + 16 .. +23
    Frag16 af[4];
    #pragma unroll
    for (int kk = 0; kk < 4; ++kk) {
        const _Float16* ap = Ah + m * FF + kk * 32 + half * 8;
        af[kk].q[0] = *reinterpret_cast<const u32x4*>(ap);
        af[kk].q[1] = *reinterpret_cast<const u32x4*>(ap + 16);
    }

    #pragma unroll
    for (int ct = 0; ct < 4; ++ct) {
        const int col = wave * 64 + ct * 16 + n;
        v8f acc = {};
        #pragma unroll
        for (int kk = 0; kk < 4; ++kk) {
            // B fragment (same K order as A): lane n holds
            // K = kk*32 + half*8 .. +7 and kk*32 + half*8 + 16 .. +23 in BT
            const _Float16* bp = BT + (size_t)col * FF + kk * 32 + half * 8;
            Frag16 bf;
            bf.q[0] = *reinterpret_cast<const u32x4*>(bp);
            bf.q[1] = *reinterpret_cast<const u32x4*>(bp + 16);
            acc = __builtin_amdgcn_wmma_f32_16x16x32_f16(
                false, af[kk].v, false, bf.v, (short)0, acc, false, false);
        }
        #pragma unroll
        for (int v = 0; v < 8; ++v)
            C[(size_t)(row0 + v + 8 * half) * OD + col] = acc[v];
    }
}

// ---------------- attention scores a_s / a_d (wave per node) ----------------
__global__ __launch_bounds__(256) void k_att(
    const float* __restrict__ lin, const float* __restrict__ att_src,
    const float* __restrict__ att_dst, float* __restrict__ a_s, float* __restrict__ a_d)
{
    const int wave = threadIdx.x >> 5;
    const int lane = threadIdx.x & 31;
    const int node = blockIdx.x * 8 + wave;
    if (node >= NN) return;
    #pragma unroll
    for (int h = 0; h < NHEAD; ++h) {
        float ps = 0.0f, pd = 0.0f;
        for (int k = lane; k < FF; k += 32) {
            float v = lin[(size_t)node * OD + h * FF + k];
            ps += v * att_src[h * FF + k];
            pd += v * att_dst[h * FF + k];
        }
        #pragma unroll
        for (int off = 16; off > 0; off >>= 1) {
            ps += __shfl_xor(ps, off, 32);
            pd += __shfl_xor(pd, off, 32);
        }
        if (lane == 0) {
            a_s[node * NHEAD + h] = ps;
            a_d[node * NHEAD + h] = pd;
        }
    }
}

// ---------------- segment max over dst (ordered-float atomicMax) ----------------
__global__ void k_edge_max(const int* __restrict__ ei, const float* __restrict__ a_s,
                           const float* __restrict__ a_d, unsigned* __restrict__ mkey)
{
    int i = blockIdx.x * blockDim.x + threadIdx.x;
    if (i >= EE * NHEAD) return;
    int e = i >> 2, h = i & 3;
    int s = ei[e], d = ei[EE + e];
    float v = leaky02(a_s[s * NHEAD + h] + a_d[d * NHEAD + h]);
    atomicMax(&mkey[d * NHEAD + h], fkey(v));
}

// ---------------- exp(e - max) + segment sum ----------------
__global__ void k_edge_exp(const int* __restrict__ ei, const float* __restrict__ a_s,
                           const float* __restrict__ a_d, const unsigned* __restrict__ mkey,
                           float* __restrict__ eedge, float* __restrict__ ssum)
{
    int i = blockIdx.x * blockDim.x + threadIdx.x;
    if (i >= EE * NHEAD) return;
    int e = i >> 2, h = i & 3;
    int s = ei[e], d = ei[EE + e];
    float v  = leaky02(a_s[s * NHEAD + h] + a_d[d * NHEAD + h]);
    float ex = expf(v - funkey(mkey[d * NHEAD + h]));
    eedge[i] = ex;
    atomicAdd(&ssum[d * NHEAD + h], ex);
}

// ---------------- weighted scatter: gat_out[dst] += (alpha/4) * lin[src] --------
// wave per edge; lanes cover 128 features (4 each)
__global__ __launch_bounds__(256) void k_edge_acc(
    const int* __restrict__ ei, const float* __restrict__ eedge,
    const float* __restrict__ ssum, const float* __restrict__ lin,
    float* __restrict__ gout)
{
    const int wid  = (blockIdx.x * blockDim.x + threadIdx.x) >> 5;
    const int lane = threadIdx.x & 31;
    if (wid >= EE) return;
    const int s = ei[wid], d = ei[EE + wid];
    #pragma unroll
    for (int h = 0; h < NHEAD; ++h) {
        float alpha = 0.25f * eedge[wid * NHEAD + h] / (ssum[d * NHEAD + h] + 1e-16f);
        #pragma unroll
        for (int j = 0; j < 4; ++j) {
            int k = lane + j * 32;
            atomicAdd(&gout[(size_t)d * FF + k], alpha * lin[(size_t)s * OD + h * FF + k]);
        }
    }
}

// ---------------- bn2 stats: one block per feature ----------------
__global__ __launch_bounds__(256) void k_bn2_stats(
    const float* __restrict__ gout, const float* __restrict__ gat_b,
    float* __restrict__ bn2m, float* __restrict__ bn2rs)
{
    __shared__ float s1[256], s2[256];
    const int f  = blockIdx.x;
    const float bb = gat_b[f];
    float a = 0.0f, b = 0.0f;
    for (int nidx = threadIdx.x; nidx < NN; nidx += 256) {
        float v = gout[(size_t)nidx * FF + f] + bb;
        a += v; b += v * v;
    }
    s1[threadIdx.x] = a; s2[threadIdx.x] = b;
    __syncthreads();
    for (int off = 128; off > 0; off >>= 1) {
        if (threadIdx.x < off) {
            s1[threadIdx.x] += s1[threadIdx.x + off];
            s2[threadIdx.x] += s2[threadIdx.x + off];
        }
        __syncthreads();
    }
    if (threadIdx.x == 0) {
        float m  = s1[0] / (float)NN;
        float vr = s2[0] / (float)NN - m * m;
        bn2m[f]  = m;
        bn2rs[f] = rsqrtf(vr + EPSV);
    }
}

// ---------------- bn2 apply ----------------
__global__ void k_bn2_apply(const float* __restrict__ gout, const float* __restrict__ gat_b,
                            const float* __restrict__ bn2m, const float* __restrict__ bn2rs,
                            float* __restrict__ h2)
{
    int i = blockIdx.x * blockDim.x + threadIdx.x;
    if (i >= NN * FF) return;
    int f = i & (FF - 1);
    h2[i] = (gout[i] + gat_b[f] - bn2m[f]) * bn2rs[f];
}

// ---------------- kernel: gated_conv2 + bn3 stats2 ----------------
__global__ __launch_bounds__(128) void k_gc2(
    const float* __restrict__ h2,
    const float* __restrict__ w1, const float* __restrict__ b1,
    const float* __restrict__ w2, const float* __restrict__ b2,
    const float* __restrict__ w3, const float* __restrict__ b3,
    float* __restrict__ g2, float* __restrict__ st2)
{
    __shared__ float xs[HID * TT];
    __shared__ float red[HID * TT];
    const int n   = blockIdx.x;
    const int tid = threadIdx.x;
    const int c   = tid >> 3;
    const int t   = tid & 7;

    xs[tid] = h2[(size_t)n * FF + tid];
    __syncthreads();

    float p = b1[c], q = b2[c], r = b3[c];
    #pragma unroll
    for (int j = 0; j < HID; ++j) {
        const float x0 = (t > 0) ? xs[j * TT + t - 1] : 0.0f;
        const float x1 = xs[j * TT + t];
        const float x2 = (t < 7) ? xs[j * TT + t + 1] : 0.0f;
        const int   wb = (c * HID + j) * 3;
        p += w1[wb] * x0 + w1[wb + 1] * x1 + w1[wb + 2] * x2;
        q += w2[wb] * x0 + w2[wb + 1] * x1 + w2[wb + 2] * x2;
        r += w3[wb] * x0 + w3[wb + 1] * x1 + w3[wb + 2] * x2;
    }
    float g = fmaxf(p * sigmoidf_(q) + r, 0.0f);
    g2[(size_t)n * FF + tid] = g;

    red[tid] = g;
    __syncthreads();
    if (t == 0) {
        float s = 0.0f, sq = 0.0f;
        #pragma unroll
        for (int tt2 = 0; tt2 < TT; ++tt2) { float v = red[c * TT + tt2]; s += v; sq += v * v; }
        atomicAdd(&st2[c], s);
        atomicAdd(&st2[HID + c], sq);
    }
}

// ---------------- final: bn3 + residual + relu @ t=7, MLP head ----------------
__global__ void k_final(const float* __restrict__ g2, const float* __restrict__ st2,
                        const float* __restrict__ resid,
                        const float* __restrict__ ow1, const float* __restrict__ ob1,
                        const float* __restrict__ ow2, const float* __restrict__ ob2,
                        float* __restrict__ out)
{
    int n = blockIdx.x * blockDim.x + threadIdx.x;
    if (n >= NN) return;
    const float cnt = (float)(NN * TT);
    float z[HID];
    #pragma unroll
    for (int c = 0; c < HID; ++c) {
        float m  = st2[c] / cnt;
        float vr = st2[HID + c] / cnt - m * m;
        float v  = (g2[(size_t)n * FF + c * TT + 7] - m) * rsqrtf(vr + EPSV)
                 + resid[(size_t)n * FF + c * TT + 7];
        z[c] = fmaxf(v, 0.0f);
    }
    float acc = ob2[0];
    for (int j = 0; j < 64; ++j) {
        float zj = ob1[j];
        #pragma unroll
        for (int c = 0; c < HID; ++c) zj += z[c] * ow1[c * 64 + j];
        acc += fmaxf(zj, 0.0f) * ow2[j];
    }
    out[n] = acc;
}

// =====================================================================
extern "C" void kernel_launch(void* const* d_in, const int* in_sizes, int n_in,
                              void* d_out, int out_size, void* d_ws, size_t ws_size,
                              hipStream_t stream)
{
    const float* X       = (const float*)d_in[0];
    const int*   ei      = (const int*)  d_in[1];
    const float* w_in    = (const float*)d_in[2];
    const float* b_in    = (const float*)d_in[3];
    const float* res_w   = (const float*)d_in[4];
    const float* res_b   = (const float*)d_in[5];
    const float* tg1_w1  = (const float*)d_in[6];
    const float* tg1_b1  = (const float*)d_in[7];
    const float* tg1_w2  = (const float*)d_in[8];
    const float* tg1_b2  = (const float*)d_in[9];
    const float* tg1_w3  = (const float*)d_in[10];
    const float* tg1_b3  = (const float*)d_in[11];
    const float* gat_w   = (const float*)d_in[12];
    const float* att_src = (const float*)d_in[13];
    const float* att_dst = (const float*)d_in[14];
    const float* gat_b   = (const float*)d_in[15];
    const float* tg2_w1  = (const float*)d_in[16];
    const float* tg2_b1  = (const float*)d_in[17];
    const float* tg2_w2  = (const float*)d_in[18];
    const float* tg2_b2  = (const float*)d_in[19];
    const float* tg2_w3  = (const float*)d_in[20];
    const float* tg2_b3  = (const float*)d_in[21];
    const float* out_w1  = (const float*)d_in[22];
    const float* out_b1  = (const float*)d_in[23];
    const float* out_w2  = (const float*)d_in[24];
    const float* out_b2  = (const float*)d_in[25];
    float* out = (float*)d_out;

    // ---- workspace carve-up (float units) ----
    float* ws = (float*)d_ws;
    size_t o = 0;
    float*    resid = ws + o; o += (size_t)NN * FF;
    float*    h1    = ws + o; o += (size_t)NN * FF;
    float*    lin   = ws + o; o += (size_t)NN * OD;
    float*    gout  = ws + o; o += (size_t)NN * FF;
    float*    h2    = ws + o; o += (size_t)NN * FF;
    float*    g2    = ws + o; o += (size_t)NN * FF;
    float*    a_s   = ws + o; o += (size_t)NN * NHEAD;
    float*    a_d   = ws + o; o += (size_t)NN * NHEAD;
    float*    ssum  = ws + o; o += (size_t)NN * NHEAD;
    unsigned* mkey  = (unsigned*)(ws + o); o += (size_t)NN * NHEAD;
    float*    eedge = ws + o; o += (size_t)EE * NHEAD;
    float*    st1   = ws + o; o += 32;
    float*    st2   = ws + o; o += 32;
    float*    bn2m  = ws + o; o += FF;
    float*    bn2rs = ws + o; o += FF;
    _Float16* A16   = (_Float16*)(ws + o); o += (size_t)NN * FF / 2;
    _Float16* W16T  = (_Float16*)(ws + o); o += (size_t)FF * OD / 2;

    // ---- zero accumulators (every call: deterministic, graph-safe) ----
    k_zero<<<1, 64, 0, stream>>>(st1, 64);                              // st1+st2
    k_zero<<<(NN * NHEAD + 255) / 256, 256, 0, stream>>>(ssum, NN * NHEAD);
    k_zero<<<(NN * NHEAD + 255) / 256, 256, 0, stream>>>((float*)mkey, NN * NHEAD);
    k_zero<<<(NN * FF + 255) / 256, 256, 0, stream>>>(gout, NN * FF);

    // ---- pipeline ----
    k_node1<<<NN, 128, 0, stream>>>(X, w_in, b_in, res_w, res_b,
                                    tg1_w1, tg1_b1, tg1_w2, tg1_b2, tg1_w3, tg1_b3,
                                    resid, h1, st1);
    k_bn3a<<<(NN * FF + 255) / 256, 256, 0, stream>>>(h1, st1, A16);
    k_wcvtT<<<(FF * OD + 255) / 256, 256, 0, stream>>>(gat_w, W16T);

    k_gemm<<<NN / 16, 256, 0, stream>>>(A16, W16T, lin);                // v_wmma path

    k_att<<<(NN + 7) / 8, 256, 0, stream>>>(lin, att_src, att_dst, a_s, a_d);
    k_edge_max<<<(EE * NHEAD + 255) / 256, 256, 0, stream>>>(ei, a_s, a_d, mkey);
    k_edge_exp<<<(EE * NHEAD + 255) / 256, 256, 0, stream>>>(ei, a_s, a_d, mkey, eedge, ssum);
    k_edge_acc<<<(EE + 7) / 8, 256, 0, stream>>>(ei, eedge, ssum, lin, gout);

    k_bn2_stats<<<FF, 256, 0, stream>>>(gout, gat_b, bn2m, bn2rs);
    k_bn2_apply<<<(NN * FF + 255) / 256, 256, 0, stream>>>(gout, gat_b, bn2m, bn2rs, h2);

    k_gc2<<<NN, 128, 0, stream>>>(h2, tg2_w1, tg2_b1, tg2_w2, tg2_b2, tg2_w3, tg2_b3,
                                  g2, st2);
    k_final<<<(NN + 255) / 256, 256, 0, stream>>>(g2, st2, resid,
                                                  out_w1, out_b1, out_w2, out_b2, out);
}
